// Policynet_66838281061057
// MI455X (gfx1250) — matrix operand order repaired and checked by reading, not compiled, not run on hardware
//
#include <hip/hip_runtime.h>
#include <hip/hip_bf16.h>
#include <math.h>

typedef __attribute__((ext_vector_type(16))) _Float16 v16h;
typedef __attribute__((ext_vector_type(8)))  _Float16 v8h;
typedef __attribute__((ext_vector_type(8)))  float    v8f;
typedef __attribute__((ext_vector_type(4)))  float    v4f;

#define Bn 65536
#define Sn 64
#define Hn 256
#define An 16
#define HPAD 272              // padded row stride in halfs (544B: 32B aligned, 8-bank shift/row)
#define LOG2PI 1.8377866f

// d_ws f16 layout offsets (elements)
#define OFF_W1  0             // [256][64]
#define OFF_W2  16384         // [256][256]
#define OFF_WMU 81920         // [16][256]
#define OFF_WLS 86016         // [16][256]
#define NW_TOT  90112

__device__ __forceinline__ v8f wmma16(v16h a, v16h b, v8f c) {
  return __builtin_amdgcn_wmma_f32_16x16x32_f16(
      /*neg_a=*/false, a, /*neg_b=*/false, b,
      /*c_mod=*/(short)0, c, /*reuse_a=*/false, /*reuse_b=*/false);
}

__global__ void prep_weights(const float* __restrict__ W1, const float* __restrict__ W2,
                             const float* __restrict__ Wmu, const float* __restrict__ Wls,
                             _Float16* __restrict__ wh) {
  int i = blockIdx.x * blockDim.x + threadIdx.x;
  if (i >= NW_TOT) return;
  float v;
  if      (i < OFF_W2)  v = W1[i - OFF_W1];
  else if (i < OFF_WMU) v = W2[i - OFF_W2];
  else if (i < OFF_WLS) v = Wmu[i - OFF_WMU];
  else                  v = Wls[i - OFF_WLS];
  wh[i] = (_Float16)v;
}

__launch_bounds__(256)
__global__ void policy_fwd(const float* __restrict__ x, const float* __restrict__ eps,
                           const float* __restrict__ b1, const float* __restrict__ b2,
                           const float* __restrict__ bmu, const float* __restrict__ bls,
                           const _Float16* __restrict__ wh, float* __restrict__ out) {
  // h1/h2: per-wave transpose staging (no cross-wave sharing).
  // W2s: block-shared copy of W2 (f16, padded rows), filled once via async-to-LDS.
  __shared__ __align__(32) _Float16 h1[128 * HPAD];
  __shared__ __align__(32) _Float16 h2[128 * HPAD];
  __shared__ __align__(32) _Float16 W2s[Hn * HPAD];

  const _Float16* W1h  = wh + OFF_W1;
  const _Float16* W2h  = wh + OFF_W2;
  const _Float16* Wmuh = wh + OFF_WMU;
  const _Float16* Wlsh = wh + OFF_WLS;

  const int tid  = threadIdx.x;
  const int wave = tid >> 5;
  const int lane = tid & 31;
  const int ln16 = lane & 15;   // N index / A-row-within-tile
  const int hlf  = lane >> 4;   // which K-chunk this lane carries

  // ---- Kick off async staging of W2 (256x256 f16 -> padded 256x272) into LDS ----
  // 8192 16B chunks, 256 threads -> 32 chunks each. Tracked by ASYNCcnt; the wait
  // is deferred to after layer 1 so the DMA runs underneath the first GEMM.
  {
    #pragma unroll 4
    for (int i = 0; i < 32; ++i) {
      const int idx = i * 256 + tid;
      const int row = idx >> 5;        // 32 chunks of 16B per 512B row
      const int co  = idx & 31;
      unsigned lds_addr = (unsigned)(size_t)(&W2s[0]) + (unsigned)(row * (HPAD * 2) + co * 16);
      unsigned long long gaddr =
          (unsigned long long)(size_t)(W2h) + (unsigned long long)(row * (Hn * 2) + co * 16);
      asm volatile("global_load_async_to_lds_b128 %0, %1, off"
                   :: "v"(lds_addr), "v"(gaddr) : "memory");
    }
  }

  const int  lrow = wave * 16 + ln16;                    // local A row (0..127)
  const long grow = (long)blockIdx.x * 128 + lrow;       // global A row

  // ---------------- Layer 1: x[64] -> h1[256], relu ----------------
  // A fragments (K = 64 -> two k-steps of 32), built from fp32 x.
  v16h a1[2];
  {
    const float* xr = x + grow * Sn;
    #pragma unroll
    for (int kk = 0; kk < 2; ++kk) {
      const int k0 = kk * 32 + hlf * 16;      // elems 0..7  : K = k0..k0+7
      v4f p0 = *(const v4f*)(xr + k0);        // elems 8..15 : K = k0+8..k0+15
      v4f p1 = *(const v4f*)(xr + k0 + 4);
      v4f q0 = *(const v4f*)(xr + k0 + 8);
      v4f q1 = *(const v4f*)(xr + k0 + 12);
      v16h a;
      #pragma unroll
      for (int e = 0; e < 4; ++e) {
        a[e]      = (_Float16)p0[e];
        a[4 + e]  = (_Float16)p1[e];
        a[8 + e]  = (_Float16)q0[e];
        a[12 + e] = (_Float16)q1[e];
      }
      a1[kk] = a;
    }
  }

  for (int nt = 0; nt < 16; ++nt) {
    const float bb = b1[nt * 16 + ln16];
    v8f acc = {bb, bb, bb, bb, bb, bb, bb, bb};
    #pragma unroll
    for (int kk = 0; kk < 2; ++kk) {
      // B fragment = W1 row (nt*16+ln16), 16 contiguous halfs (32B) -> WGP$-resident.
      v16h bf = *(const v16h*)(W1h + (nt * 16 + ln16) * Sn + kk * 32 + hlf * 16);
      acc = wmma16(a1[kk], bf, acc);
    }
    const int col   = nt * 16 + ln16;
    const int rbase = wave * 16 + hlf * 8;    // D layout: VGPR g -> row hlf*8+g
    #pragma unroll
    for (int g = 0; g < 8; ++g) {
      float v = acc[g] > 0.0f ? acc[g] : 0.0f;
      h1[(rbase + g) * HPAD + col] = (_Float16)v;
    }
  }

  // Drain our async copies, then make the staged W2 visible to all waves.
  asm volatile("s_wait_asynccnt 0x0" ::: "memory");
  __syncthreads();

  // ---------------- Layer 2: h1[256] -> h2[256], relu (B from LDS) ----------------
  const _Float16* aRow1 = h1 + lrow * HPAD;
  for (int ng = 0; ng < 4; ++ng) {            // 4 N-tiles per group: A reuse x4
    v8f acc[4];
    #pragma unroll
    for (int j = 0; j < 4; ++j) {
      const float bb = b2[(ng * 4 + j) * 16 + ln16];
      acc[j] = (v8f){bb, bb, bb, bb, bb, bb, bb, bb};
    }
    for (int kk = 0; kk < 8; ++kk) {
      const int k0 = kk * 32 + hlf * 16;
      v8h c0 = *(const v8h*)(aRow1 + k0);       // ds_load_b128
      v8h c1 = *(const v8h*)(aRow1 + k0 + 8);   // ds_load_b128
      v16h a;
      #pragma unroll
      for (int e = 0; e < 8; ++e) { a[e] = c0[e]; a[8 + e] = c1[e]; }
      #pragma unroll
      for (int j = 0; j < 4; ++j) {
        const int nt = ng * 4 + j;
        v16h bf = *(const v16h*)(W2s + (nt * 16 + ln16) * HPAD + kk * 32 + hlf * 16);
        acc[j] = wmma16(a, bf, acc[j]);
      }
    }
    #pragma unroll
    for (int j = 0; j < 4; ++j) {
      const int col   = (ng * 4 + j) * 16 + ln16;
      const int rbase = wave * 16 + hlf * 8;
      #pragma unroll
      for (int g = 0; g < 8; ++g) {
        float v = acc[j][g] > 0.0f ? acc[j][g] : 0.0f;
        h2[(rbase + g) * HPAD + col] = (_Float16)v;
      }
    }
  }

  // ---------------- Heads: h2[256] -> mu[16], ls[16] (B hits WGP$) ----------------
  const float bmv = bmu[ln16];
  const float blv = bls[ln16];
  v8f amu = {bmv, bmv, bmv, bmv, bmv, bmv, bmv, bmv};
  v8f als = {blv, blv, blv, blv, blv, blv, blv, blv};
  const _Float16* aRow2 = h2 + lrow * HPAD;
  for (int kk = 0; kk < 8; ++kk) {
    const int k0 = kk * 32 + hlf * 16;
    v8h c0 = *(const v8h*)(aRow2 + k0);
    v8h c1 = *(const v8h*)(aRow2 + k0 + 8);
    v16h a;
    #pragma unroll
    for (int e = 0; e < 8; ++e) { a[e] = c0[e]; a[8 + e] = c1[e]; }
    v16h bm = *(const v16h*)(Wmuh + ln16 * Hn + kk * 32 + hlf * 16);
    v16h bl = *(const v16h*)(Wlsh + ln16 * Hn + kk * 32 + hlf * 16);
    amu = wmma16(a, bm, amu);
    als = wmma16(a, bl, als);
  }

  // ---------------- Epilogue: rsample + diagonal-Gaussian logprob ----------------
  const long rowg0 = (long)blockIdx.x * 128 + wave * 16 + hlf * 8;
  const int  acol  = ln16;
  #pragma unroll
  for (int g = 0; g < 8; ++g) {
    const long row = rowg0 + g;
    float mu = tanhf(amu[g]);
    mu = -1.0f + 0.5f * (mu + 1.0f) * 2.0f;               // MIN_A + 0.5*(mu+1)*(MAX-MIN)
    const float z  = als[g];
    const float sp = (z > 20.0f) ? z : log1pf(expf(z));   // stable softplus
    const float sd = sp + 1e-6f;
    const float e  = eps[row * An + acol];
    out[row * An + acol] = mu + sd * e;
    float t = e * e + 2.0f * logf(sd);
    #pragma unroll
    for (int off = 1; off < 16; off <<= 1)                // sum over 16 cols (stays in half-wave)
      t += __shfl_xor(t, off, 32);
    if (acol == 0)
      out[(long)Bn * An + row] = -0.5f * t - 0.5f * (float)An * LOG2PI;
  }
}

extern "C" void kernel_launch(void* const* d_in, const int* in_sizes, int n_in,
                              void* d_out, int out_size, void* d_ws, size_t ws_size,
                              hipStream_t stream) {
  const float* x   = (const float*)d_in[0];
  const float* eps = (const float*)d_in[1];
  const float* W1  = (const float*)d_in[2];
  const float* b1  = (const float*)d_in[3];
  const float* W2  = (const float*)d_in[4];
  const float* b2  = (const float*)d_in[5];
  const float* Wmu = (const float*)d_in[6];
  const float* bmu = (const float*)d_in[7];
  const float* Wls = (const float*)d_in[8];
  const float* bls = (const float*)d_in[9];
  float*       out = (float*)d_out;
  _Float16*    wh  = (_Float16*)d_ws;

  prep_weights<<<(NW_TOT + 255) / 256, 256, 0, stream>>>(W1, W2, Wmu, Wls, wh);
  policy_fwd<<<Bn / 128, 256, 0, stream>>>(x, eps, b1, b2, bmu, bls, wh, out);
}
